// DreamAttention_58025008169310
// MI455X (gfx1250) — matrix operand order repaired and checked by reading, not compiled, not run on hardware
//
#include <hip/hip_runtime.h>
#include <hip/hip_bf16.h>

// ---------------------------------------------------------------------------
// DreamAttention for MI455X (gfx1250): bf16 WMMA everywhere, fp32 accum.
// Async (memory->LDS) staging for attention V tiles via ASYNCcnt path.
// ---------------------------------------------------------------------------

typedef __attribute__((ext_vector_type(16))) __bf16 v16bf;
typedef __attribute__((ext_vector_type(8)))  __bf16 v8bf;
typedef __attribute__((ext_vector_type(8)))  float  v8f;

constexpr int BB  = 2;
constexpr int LL  = 2048;
constexpr int TT  = BB * LL;       // 4096
constexpr int HID = 2048;
constexpr int HH  = 16;
constexpr int HKV = 8;
constexpr int DD  = 128;
constexpr int SS  = 1024;
// G = HH / HKV = 2

// ln(1e6)/64  (RoPE inv-freq exponent step)
constexpr float ROPE_LN_STEP = 0.21586735246819178f;
constexpr float QK_SCALE     = 0.08838834764831845f;   // 1/sqrt(128)

static __device__ inline v16bf cat8(v8bf lo, v8bf hi) {
    return __builtin_shufflevector(lo, hi, 0,1,2,3,4,5,6,7,8,9,10,11,12,13,14,15);
}
static __device__ inline v8f wmma_bf16(v16bf a, v16bf b, v8f c) {
    // D = A(16x32) * B(32x16) + C, f32 accumulate
    return __builtin_amdgcn_wmma_f32_16x16x32_bf16(false, a, false, b, (short)0, c, false, false);
}

// memory -> LDS async copy (16B per lane), tracked by ASYNCcnt
static __device__ inline void async_load_lds_b128(unsigned lds_byte_off,
                                                  unsigned long long gaddr) {
    asm volatile("global_load_async_to_lds_b128 %0, %1, off"
                 :: "v"(lds_byte_off), "v"(gaddr)
                 : "memory");
}
static __device__ inline void wait_asynccnt0() {
#if __has_builtin(__builtin_amdgcn_s_wait_asynccnt)
    __builtin_amdgcn_s_wait_asynccnt(0);
#else
    asm volatile("s_wait_asynccnt 0x0" ::: "memory");
#endif
}

// ---------------------------------------------------------------------------
// Elementwise fp32 -> bf16
// ---------------------------------------------------------------------------
__global__ void cvt_f32_bf16_kernel(const float* __restrict__ in,
                                    __bf16* __restrict__ out, int n) {
    int i = blockIdx.x * blockDim.x + threadIdx.x;
    if (i < n) out[i] = (__bf16)in[i];
}

__global__ void cvt_caches_kernel(const float* __restrict__ kin,
                                  const float* __restrict__ vin,
                                  __bf16* __restrict__ kout,
                                  __bf16* __restrict__ vout, int n) {
    int i = blockIdx.x * blockDim.x + threadIdx.x;
    if (i < n) { kout[i] = (__bf16)kin[i]; vout[i] = (__bf16)vin[i]; }
}

// ---------------------------------------------------------------------------
// Transpose W[R,C] fp32 -> WT[C,R] bf16 (LDS tiled, coalesced both ways)
// ---------------------------------------------------------------------------
__global__ void transpose_w_kernel(const float* __restrict__ W,
                                   __bf16* __restrict__ WT, int R, int C) {
    __shared__ float tile[16][17];
    int c = blockIdx.x * 16 + threadIdx.x;
    int r = blockIdx.y * 16 + threadIdx.y;
    tile[threadIdx.y][threadIdx.x] = W[(size_t)r * C + c];
    __syncthreads();
    int co = blockIdx.x * 16 + threadIdx.y;
    int ro = blockIdx.y * 16 + threadIdx.x;
    WT[(size_t)co * R + ro] = (__bf16)tile[threadIdx.x][threadIdx.y];
}

// ---------------------------------------------------------------------------
// Generic bf16 WMMA GEMM:  C[M,N] = (gather? A[g[m]]: A[m]) @ WT[N,K]^T + bias
// One wave -> 16x64 tile (1 A frag reused by 4 B frags per K-step).
// 8 waves/block tile N by 512.  Output bf16 (outb) or fp32 (outf).
// ---------------------------------------------------------------------------
__global__ __launch_bounds__(256)
void gemm_bf16_kernel(const __bf16* __restrict__ A,
                      const __bf16* __restrict__ WT,
                      const float* __restrict__ bias,
                      const int* __restrict__ gather,
                      __bf16* __restrict__ outb,
                      float* __restrict__ outf,
                      int N, int K) {
    const int wave = threadIdx.x >> 5;
    const int lane = threadIdx.x & 31;
    const int lh   = lane & 15;       // row/col within tile
    const int kh   = lane >> 4;       // K-half selector
    const int m0   = blockIdx.x * 16;
    const int nb   = (blockIdx.y * 8 + wave) * 64;

    int mrow = m0 + lh;
    int arow = gather ? gather[mrow] : mrow;
    const __bf16* ap = A + (size_t)arow * K;

    v8f acc[4] = {};
    for (int k0 = 0; k0 < K; k0 += 32) {
        // gfx1250 prefetch hint for the streams (global_prefetch_b8)
        __builtin_prefetch(ap + k0 + 256, 0, 1);
        __builtin_prefetch(WT + (size_t)(nb + lh) * K + k0 + 256, 0, 1);

        const __bf16* pa = ap + k0 + kh * 16;
        v16bf af = cat8(*(const v8bf*)pa, *(const v8bf*)(pa + 8));
#pragma unroll
        for (int j = 0; j < 4; ++j) {
            const __bf16* pb = WT + (size_t)(nb + j * 16 + lh) * K + k0 + kh * 16;
            acc[j] = wmma_bf16(af, *(const v16bf*)pb, acc[j]);
        }
    }
#pragma unroll
    for (int j = 0; j < 4; ++j) {
        int n = nb + j * 16 + lh;
        float bv = bias ? bias[n] : 0.0f;
#pragma unroll
        for (int r = 0; r < 8; ++r) {
            int m = m0 + r + kh * 8;           // C layout: lanes16-31 hold M=r+8
            float v = acc[j][r] + bv;
            if (outf) outf[(size_t)m * N + n] = v;
            else      outb[(size_t)m * N + n] = (__bf16)v;
        }
    }
}

// ---------------------------------------------------------------------------
// RoPE on q:  qlin[T, H*D] -> qr[(b*H+h)*L + l, D], scaled by 1/sqrt(D)
// ---------------------------------------------------------------------------
__global__ void rope_q_kernel(const __bf16* __restrict__ qlin,
                              const int* __restrict__ pos,
                              __bf16* __restrict__ qr) {
    int i = blockIdx.x * blockDim.x + threadIdx.x;   // T*H*64
    int j = i & 63;
    int h = (i >> 6) & (HH - 1);
    int t = i >> 10;
    if (t >= TT) return;
    float p = (float)pos[t];
    float f = p * __expf(-(float)j * ROPE_LN_STEP);
    float sn, cs; __sincosf(f, &sn, &cs);
    float x1 = (float)qlin[(size_t)t * HID + h * DD + j];
    float x2 = (float)qlin[(size_t)t * HID + h * DD + j + 64];
    int b = t >> 11, l = t & (LL - 1);
    size_t orow = ((size_t)(b * HH + h) * LL + l) * DD;
    qr[orow + j]      = (__bf16)((x1 * cs - x2 * sn) * QK_SCALE);
    qr[orow + 64 + j] = (__bf16)((x2 * cs + x1 * sn) * QK_SCALE);
}

// ---------------------------------------------------------------------------
// RoPE k + scatter salient k/v rows into the bf16 caches [T, HKV, D]
// ---------------------------------------------------------------------------
__global__ void rope_scatter_kv_kernel(const __bf16* __restrict__ kvlin,
                                       const int* __restrict__ pos,
                                       const int* __restrict__ idx,
                                       __bf16* __restrict__ kcache,
                                       __bf16* __restrict__ vcache) {
    int i = blockIdx.x * blockDim.x + threadIdx.x;   // S*HKV*64
    int j   = i & 63;
    int kvh = (i >> 6) & (HKV - 1);
    int s   = i >> 9;
    if (s >= SS) return;
    int t = idx[s];
    float p = (float)pos[t];
    float f = p * __expf(-(float)j * ROPE_LN_STEP);
    float sn, cs; __sincosf(f, &sn, &cs);
    float k1 = (float)kvlin[(size_t)s * HID + kvh * DD + j];
    float k2 = (float)kvlin[(size_t)s * HID + kvh * DD + j + 64];
    size_t row = ((size_t)t * HKV + kvh) * DD;
    kcache[row + j]      = (__bf16)(k1 * cs - k2 * sn);
    kcache[row + 64 + j] = (__bf16)(k2 * cs + k1 * sn);
    vcache[row + j]      = kvlin[(size_t)s * HID + HKV * DD + kvh * DD + j];
    vcache[row + 64 + j] = kvlin[(size_t)s * HID + HKV * DD + kvh * DD + j + 64];
}

// ---------------------------------------------------------------------------
// Attention: one block per (b, h, q-tile of 16). 8 waves.
//   Phase 1: S = Q K^T via WMMA -> bf16 scores in LDS (16 x 2048)
//   Phase 2: per-row softmax (wave32 shfl_xor reductions)
//   Phase 3: O = P V via WMMA; V tiles staged into LDS with
//            GLOBAL_LOAD_ASYNC_TO_LDS_B128 (ASYNCcnt path, no VGPR roundtrip)
// Dynamic LDS: 16*2048*2 (scores) + 32*128*2 (V tile) = 73728 B
// ---------------------------------------------------------------------------
__global__ __launch_bounds__(256)
void attn_kernel(const __bf16* __restrict__ q,
                 const __bf16* __restrict__ kc,
                 const __bf16* __restrict__ vc,
                 __bf16* __restrict__ ob) {
    extern __shared__ char smem[];
    __bf16* sc = (__bf16*)smem;                       // [16][LL] scores/probs
    constexpr unsigned VT_BASE = 16 * LL * 2;         // byte offset of V tile
    __bf16* vt = (__bf16*)(smem + VT_BASE);           // [32 s][128 d] row-major

    int bx  = blockIdx.x;
    int qt  = bx & 127;
    int h   = (bx >> 7) & (HH - 1);
    int b   = bx >> 11;
    int kvh = h >> 1;                                  // G = 2

    int tid = threadIdx.x, wave = tid >> 5, lane = tid & 31;
    int lh = lane & 15, kh = lane >> 4;

    // Q fragments for this 16-row tile (shared by all waves; hits L0/L2)
    const __bf16* qrow = q + ((size_t)(b * HH + h) * LL + qt * 16 + lh) * DD;
    v16bf aq[4];
#pragma unroll
    for (int ks = 0; ks < 4; ++ks) {
        const __bf16* p = qrow + ks * 32 + kh * 16;
        aq[ks] = cat8(*(const v8bf*)p, *(const v8bf*)(p + 8));
    }

    // ---- Phase 1: scores (each wave owns 256 key positions = 16 tiles) ----
    for (int st = wave * 16; st < wave * 16 + 16; ++st) {
        int s = st * 16 + lh;
        const __bf16* krow = kc + ((size_t)(b * LL + s) * HKV + kvh) * DD;
        v8f c = {};
#pragma unroll
        for (int ks = 0; ks < 4; ++ks)
            c = wmma_bf16(aq[ks], *(const v16bf*)(krow + ks * 32 + kh * 16), c);
#pragma unroll
        for (int r = 0; r < 8; ++r)
            sc[(size_t)(r + kh * 8) * LL + st * 16 + lh] = (__bf16)c[r];
    }
    __syncthreads();

    // ---- Phase 2: softmax over the full 2048 keys (non-causal) ----
#pragma unroll
    for (int rr = 0; rr < 2; ++rr) {
        __bf16* row = sc + (size_t)(wave + rr * 8) * LL;
        float mx = -3.0e38f;
        for (int c0 = lane; c0 < LL; c0 += 32) mx = fmaxf(mx, (float)row[c0]);
        for (int off = 16; off; off >>= 1) mx = fmaxf(mx, __shfl_xor(mx, off, 32));
        float sum = 0.0f;
        for (int c0 = lane; c0 < LL; c0 += 32) {
            float e = __expf((float)row[c0] - mx);
            sum += e;
            row[c0] = (__bf16)e;
        }
        for (int off = 16; off; off >>= 1) sum += __shfl_xor(sum, off, 32);
        float inv = 1.0f / sum;
        for (int c0 = lane; c0 < LL; c0 += 32)
            row[c0] = (__bf16)((float)row[c0] * inv);
    }
    __syncthreads();

    // ---- Phase 3: O = P @ V (wave owns 16 d-columns) ----
    int dcol = wave * 16 + lh;
    v8f oacc = {};
    for (int st = 0; st < LL / 32; ++st) {
        int s0 = st * 32;
        // Async-stage the 32x128 V tile row-major into LDS:
        // 512 chunks of 16B; 2 chunks per thread; no VGPR data roundtrip.
#pragma unroll
        for (int p = 0; p < 2; ++p) {
            int idx = p * 256 + tid;
            int sr  = idx >> 4;             // 0..31 key row in tile
            int dc  = (idx & 15) * 8;       // bf16 column chunk
            unsigned lds_off = VT_BASE + (unsigned)(sr * DD + dc) * 2u;
            const __bf16* gsrc =
                vc + ((size_t)(b * LL + s0 + sr) * HKV + kvh) * DD + dc;
            async_load_lds_b128(lds_off, (unsigned long long)(const void*)gsrc);
        }
        wait_asynccnt0();
        __syncthreads();

        // A fragment: probs straight out of LDS (contiguous b128 ds loads)
        const __bf16* pr = sc + (size_t)lh * LL + s0 + kh * 16;
        v16bf pa = cat8(*(const v8bf*)pr, *(const v8bf*)(pr + 8));

        // B fragment: V^T view of the row-major LDS tile
        v16bf pb;
#pragma unroll
        for (int e = 0; e < 16; ++e)
            pb[e] = vt[(size_t)(kh * 16 + e) * DD + dcol];

        oacc = wmma_bf16(pa, pb, oacc);
        __syncthreads();
    }
#pragma unroll
    for (int r = 0; r < 8; ++r) {
        int m = r + kh * 8;
        ob[(size_t)(b * LL + qt * 16 + m) * (HH * DD) + h * DD + dcol] = (__bf16)oacc[r];
    }
}

// ---------------------------------------------------------------------------
// Host orchestration
// ---------------------------------------------------------------------------
extern "C" void kernel_launch(void* const* d_in, const int* in_sizes, int n_in,
                              void* d_out, int out_size, void* d_ws, size_t ws_size,
                              hipStream_t stream) {
    (void)in_sizes; (void)n_in; (void)out_size; (void)ws_size;

    const int*   positions = (const int*)  d_in[0];
    const float* hidden    = (const float*)d_in[1];
    const int*   idx_sal   = (const int*)  d_in[2];
    const float* k_prev    = (const float*)d_in[3];
    const float* v_prev    = (const float*)d_in[4];
    const float* Wq        = (const float*)d_in[5];
    const float* bq        = (const float*)d_in[6];
    const float* Wkv       = (const float*)d_in[7];
    const float* bkv       = (const float*)d_in[8];
    const float* Wo        = (const float*)d_in[9];
    float* out = (float*)d_out;

    constexpr size_t MB = 1ull << 20;
    char* ws = (char*)d_ws;
    __bf16* hs_b    = (__bf16*)(ws + 0 * MB);    // [T, HID]          16 MB
    __bf16* WqT     = (__bf16*)(ws + 16 * MB);   // [2048, 2048]       8 MB
    __bf16* WkvT    = (__bf16*)(ws + 24 * MB);   // [2048, 2048]       8 MB
    __bf16* WoT     = (__bf16*)(ws + 32 * MB);   // [2048, 2048]       8 MB
    __bf16* q_lin   = (__bf16*)(ws + 40 * MB);   // [T, H*D]          16 MB
    __bf16* q_rope  = (__bf16*)(ws + 56 * MB);   // [B,H,L,D]         16 MB
    __bf16* kv_lin  = (__bf16*)(ws + 72 * MB);   // [S, 2*HKV*D]       4 MB
    __bf16* k_cache = (__bf16*)(ws + 76 * MB);   // [T, HKV, D]        8 MB
    __bf16* v_cache = (__bf16*)(ws + 84 * MB);   // [T, HKV, D]        8 MB
    __bf16* o_b     = (__bf16*)(ws + 92 * MB);   // [T, H*D]          16 MB

    // 1) convert hidden states + caches to bf16; transpose weights to [N,K] bf16
    {
        int n = TT * HID;
        cvt_f32_bf16_kernel<<<(n + 255) / 256, 256, 0, stream>>>(hidden, hs_b, n);
    }
    {
        dim3 g(HID / 16, HID / 16), blk(16, 16);
        transpose_w_kernel<<<g, blk, 0, stream>>>(Wq,  WqT,  HID, HH * DD);
        transpose_w_kernel<<<g, blk, 0, stream>>>(Wkv, WkvT, HID, 2 * HKV * DD);
        transpose_w_kernel<<<g, blk, 0, stream>>>(Wo,  WoT,  HH * DD, HID);
    }
    {
        int n = TT * HKV * DD;
        cvt_caches_kernel<<<(n + 255) / 256, 256, 0, stream>>>(k_prev, v_prev,
                                                               k_cache, v_cache, n);
    }

    // 2) QKV projections (WMMA GEMMs)
    gemm_bf16_kernel<<<dim3(TT / 16, (HH * DD) / 512), 256, 0, stream>>>(
        hs_b, WqT, bq, nullptr, q_lin, nullptr, HH * DD, HID);
    gemm_bf16_kernel<<<dim3(SS / 16, (2 * HKV * DD) / 512), 256, 0, stream>>>(
        hs_b, WkvT, bkv, idx_sal, kv_lin, nullptr, 2 * HKV * DD, HID);

    // 3) RoPE + cache scatter
    {
        int n = TT * HH * 64;
        rope_q_kernel<<<(n + 255) / 256, 256, 0, stream>>>(q_lin, positions, q_rope);
    }
    {
        int n = SS * HKV * 64;
        rope_scatter_kv_kernel<<<(n + 255) / 256, 256, 0, stream>>>(
            kv_lin, positions, idx_sal, k_cache, v_cache);
    }

    // 4) attention (WMMA QK^T -> softmax -> async-staged WMMA PV)
    {
        int blocks = BB * HH * (LL / 16);                 // 4096
        size_t smem = (size_t)16 * LL * 2 + 32 * DD * 2;  // 73728 B
        attn_kernel<<<blocks, 256, smem, stream>>>(q_rope, k_cache, v_cache, o_b);
    }

    // 5) output projection, fp32 result
    gemm_bf16_kernel<<<dim3(TT / 16, HID / 512), 256, 0, stream>>>(
        o_b, WoT, nullptr, nullptr, nullptr, out, HID, HH * DD);
}
